// KalmanBlock_82497731821665
// MI455X (gfx1250) — matrix operand order repaired and checked by reading, not compiled, not run on hardware
//
#include <hip/hip_runtime.h>
#include <hip/hip_bf16.h>
#include <math.h>

// ---------------------------------------------------------------------------
// KalmanBlock for MI455X (gfx1250): bf16 WMMA everywhere.
//   B=16 batch rows == WMMA M dimension -> the whole recurrent scan runs as
//   16x16x32 v_wmma tiles inside ONE workgroup (16 waves) with state in LDS
//   and bf16 weights resident in the 192MB L2. Weight (B-matrix) tiles are
//   streamed through per-wave LDS double buffers with the CDNA5 async
//   global->LDS path (ASYNCcnt) when the toolchain exposes the builtins.
// ---------------------------------------------------------------------------

#define Bb 16
#define Tt 1024
#define Ee 1024
#define Ss 256
#define Dd 512
#define Gg 128

typedef __attribute__((ext_vector_type(16))) __bf16 v16bf;
typedef __attribute__((ext_vector_type(8)))  __bf16 v8bf;
typedef __attribute__((ext_vector_type(8)))  float  v8f;
typedef __attribute__((ext_vector_type(4)))  float  v4f;
typedef __attribute__((ext_vector_type(4)))  int    v4i;

#define WMMA_BF16(a, b, c) \
  __builtin_amdgcn_wmma_f32_16x16x32_bf16(false, (a), false, (b), (short)0, (c), false, false)

#if defined(__AMDGCN__) && __has_builtin(__builtin_amdgcn_global_load_async_to_lds_b128) && \
    __has_builtin(__builtin_amdgcn_s_wait_asynccnt)
#define HAVE_ASYNC_LDS 1
#else
#define HAVE_ASYNC_LDS 0
#endif

static __device__ __forceinline__ v8f zero8() {
  v8f z;
#pragma unroll
  for (int i = 0; i < 8; ++i) z[i] = 0.0f;
  return z;
}

#if HAVE_ASYNC_LDS
// 16B per lane, global -> LDS, tracked by ASYNCcnt (global_load_async_to_lds_b128)
static __device__ __forceinline__ void async_copy16(const void* g, void* l) {
  __attribute__((address_space(1))) void* gv =
      (__attribute__((address_space(1))) void*)(void*)g;
  __attribute__((address_space(3))) void* lv =
      (__attribute__((address_space(3))) void*)l;
  __builtin_amdgcn_global_load_async_to_lds_b128(
      (__attribute__((address_space(1))) v4i*)gv,
      (__attribute__((address_space(3))) v4i*)lv, 0, 0);
}
#endif

// A fragment (16x32 bf16): lanes 0-15 row=lane K=0..15;
// lanes 16-31 row=lane-16 K=16..31. Source f32, converted on the fly.
static __device__ __forceinline__ v16bf frag_a_f32(const float* base, size_t rstride,
                                                   int kb, int lane) {
  const int row = lane & 15, half = lane >> 4;
  const float* p = base + (size_t)row * rstride + kb + half * 16;
  v4f a0 = *(const v4f*)(p);
  v4f a1 = *(const v4f*)(p + 4);
  v4f b0 = *(const v4f*)(p + 8);
  v4f b1 = *(const v4f*)(p + 12);
  v16bf f;
#pragma unroll
  for (int i = 0; i < 4; ++i) {
    f[i]      = (__bf16)a0[i];
    f[4 + i]  = (__bf16)a1[i];
    f[8 + i]  = (__bf16)b0[i];
    f[12 + i] = (__bf16)b1[i];
  }
  return f;
}

// A fragment from a bf16 row-major [16, ld] matrix (LDS staging buffers).
static __device__ __forceinline__ v16bf frag_a_bf16(const __bf16* S, int ld,
                                                    int kb, int lane) {
  const int row = lane & 15, half = lane >> 4;
  const __bf16* p = S + (size_t)row * ld + kb + half * 16;
  v8bf lo = *(const v8bf*)(p);
  v8bf hi = *(const v8bf*)(p + 8);
  v16bf f;
#pragma unroll
  for (int i = 0; i < 8; ++i) { f[i] = lo[i]; f[8 + i] = hi[i]; }
  return f;
}

// B fragment (32x16 bf16): lanes 0-15 hold K=0..15 for col=lane, lanes 16-31
// hold K=16..31. Weights are stored [N][K] so a lane reads 32 contiguous bytes.
static __device__ __forceinline__ v16bf frag_b_bf16(const __bf16* Bt, int K,
                                                    int kb, int lane) {
  const int col = lane & 15, half = lane >> 4;
  const __bf16* p = Bt + (size_t)col * K + kb + half * 16;
  v8bf lo = *(const v8bf*)(p);
  v8bf hi = *(const v8bf*)(p + 8);
  v16bf f;
#pragma unroll
  for (int i = 0; i < 8; ++i) { f[i] = lo[i]; f[8 + i] = hi[i]; }
  return f;
}

// ---------------------------------------------------------------------------
// WMMA K-loop over a [N][K]-stored bf16 B matrix, A fragments supplied by a
// functor. When available, B tiles are streamed via async global->LDS double
// buffering (wave-private 2x1KB buffers, s_wait_asynccnt pipeline).
// ---------------------------------------------------------------------------
template <typename AF>
static __device__ __forceinline__ v8f wmma_kloop(AF af, const __bf16* Bcol, int K,
                                                 char* b0, char* b1, int lane, v8f acc) {
#if HAVE_ASYNC_LDS
  const int col = lane & 15, half = lane >> 4;
  const char* g = (const char*)(Bcol + (size_t)col * K + half * 16);
  const int lo = lane * 32;
  async_copy16(g, b0 + lo);
  async_copy16(g + 16, b0 + lo + 16);
  for (int kb = 0; kb < K; kb += 32) {
    char* cur = ((kb >> 5) & 1) ? b1 : b0;
    if (kb + 32 < K) {
      char* nxt = ((kb >> 5) & 1) ? b0 : b1;
      const char* gn = g + (size_t)(kb + 32) * 2;
      async_copy16(gn, nxt + lo);
      async_copy16(gn + 16, nxt + lo + 16);
      __builtin_amdgcn_s_wait_asynccnt(2);   // current buffer's 2 ops retired
    } else {
      __builtin_amdgcn_s_wait_asynccnt(0);   // drain
    }
    const __bf16* p = (const __bf16*)(cur + lo);
    v8bf blo = *(const v8bf*)(p);
    v8bf bhi = *(const v8bf*)(p + 8);
    v16bf b;
#pragma unroll
    for (int i = 0; i < 8; ++i) { b[i] = blo[i]; b[8 + i] = bhi[i]; }
    acc = WMMA_BF16(af(kb), b, acc);
  }
#else
  (void)b0; (void)b1;
  for (int kb = 0; kb < K; kb += 32) {
    __builtin_prefetch(Bcol + (size_t)(lane & 15) * K + kb + 128, 0, 1);
    acc = WMMA_BF16(af(kb), frag_b_bf16(Bcol, K, kb, lane), acc);
  }
#endif
  return acc;
}

static __device__ __forceinline__ float gelu_tanh(float x) {
  float x3 = x * x * x;
  return 0.5f * x * (1.0f + tanhf(0.7978845608028654f * (x + 0.044715f * x3)));
}
static __device__ __forceinline__ float sigmoidf_(float x) {
  return 1.0f / (1.0f + expf(-x));
}
static __device__ __forceinline__ float clampf(float x, float lo, float hi) {
  return fminf(fmaxf(x, lo), hi);
}

// ---------------------------------------------------------------------------
// Elementwise prep kernels
// ---------------------------------------------------------------------------
__global__ void k_cast(const float* __restrict__ s, __bf16* __restrict__ d, int n) {
  int i = blockIdx.x * blockDim.x + threadIdx.x;
  if (i < n) d[i] = (__bf16)s[i];
}

// dst[c*R + r] = src[r*C + c]  (store math-B [K,N] as [N][K] bf16)
__global__ void k_cast_t(const float* __restrict__ s, __bf16* __restrict__ d,
                         int R, int C) {
  int i = blockIdx.x * blockDim.x + threadIdx.x;
  if (i < R * C) {
    int r = i / C, c = i - r * C;
    d[(size_t)c * R + r] = (__bf16)s[i];
  }
}

__global__ void k_prep(const float* __restrict__ Q, const float* __restrict__ Rn,
                       float* __restrict__ qsp, float* __restrict__ reff) {
  __shared__ float red[512];
  int tid = threadIdx.x;
  float x = Rn[tid];
  red[tid] = (x > 20.0f) ? x : log1pf(expf(x));
  __syncthreads();
  for (int s = 256; s > 0; s >>= 1) {
    if (tid < s) red[tid] += red[tid + s];
    __syncthreads();
  }
  if (tid == 0) reff[0] = red[0] / (float)Dd;
  if (tid < Ss) {
    float q = Q[tid];
    qsp[tid] = (q > 20.0f) ? q : log1pf(expf(q));
  }
}

// ---------------------------------------------------------------------------
// Generic WMMA GEMM: C[M,N] = act(A_f32[M,K] @ B_bf16[N][K] + bias) (+resid)
// block = 256 thr = 8 waves; each wave owns one 16x16 tile; grid covers M,N.
// ---------------------------------------------------------------------------
__global__ __launch_bounds__(256)
void k_gemm_bf16(const float* __restrict__ A, const __bf16* __restrict__ Bt,
                 const float* __restrict__ bias, const float* __restrict__ resid,
                 float* __restrict__ C, int M, int N, int K, int act) {
#if HAVE_ASYNC_LDS
  __shared__ __align__(16) char stage[8][2][1024];
#endif
  const int lane = threadIdx.x & 31;
  const int wave = threadIdx.x >> 5;
  const int m0 = blockIdx.y << 4;
  const int n0 = (blockIdx.x * 8 + wave) << 4;
  if (m0 >= M || n0 >= N) return;
  const float*  Arow = A + (size_t)m0 * K;
  const __bf16* Bcol = Bt + (size_t)n0 * K;
  char *sb0 = nullptr, *sb1 = nullptr;
#if HAVE_ASYNC_LDS
  sb0 = &stage[wave][0][0];
  sb1 = &stage[wave][1][0];
#endif
  v8f acc = wmma_kloop(
      [&](int kb) { return frag_a_f32(Arow, (size_t)K, kb, lane); },
      Bcol, K, sb0, sb1, lane, zero8());

  const int col = lane & 15, rb = (lane >> 4) * 8;
  const int n = n0 + col;
  const float bn = bias ? bias[n] : 0.0f;
#pragma unroll
  for (int i = 0; i < 8; ++i) {
    int m = m0 + rb + i;
    float v = acc[i] + bn;
    if (act) v = gelu_tanh(v);
    if (resid) v += resid[(size_t)m * N + n];
    C[(size_t)m * N + n] = v;
  }
}

// ---------------------------------------------------------------------------
// The recurrent Kalman+GRU scan. ONE workgroup, 16 waves. M=16 (=batch) per
// WMMA tile; 16 waves cover all N tiles of each phase. State lives in LDS.
// ---------------------------------------------------------------------------
#define SCAN_THREADS 512
#define SCAN_SMEM 138784  // 106016 state bytes + 16 waves * 2 * 1KB async staging

__global__ __launch_bounds__(SCAN_THREADS)
void k_scan(const float* __restrict__ u, float* __restrict__ xs,
            const __bf16* __restrict__ Ab,   const __bf16* __restrict__ WstT,
            const __bf16* __restrict__ Hb,   const __bf16* __restrict__ Ht,
            const __bf16* __restrict__ Wz,   const __bf16* __restrict__ Wr,
            const __bf16* __restrict__ Wh,   const __bf16* __restrict__ WoutT,
            const float* __restrict__ b_state, const float* __restrict__ b_z,
            const float* __restrict__ b_r,     const float* __restrict__ b_h,
            const float* __restrict__ qsp,     const float* __restrict__ reff) {
  extern __shared__ char smem[];
  float* sXe  = (float*)smem;          // [16,256] x_est -> x_post -> x_final
  float* sXp  = sXe + 16 * Ss;         // [16,256] x_pred
  float* sP   = sXp + 16 * Ss;         // [16,256] covariance
  float* sH   = sP  + 16 * Ss;         // [16,128] GRU hidden
  float* sZ   = sH  + 16 * Gg;         // [16,128] z gate
  float* sR   = sZ  + 16 * Gg;         // [16,128] r gate
  float* sBst = sR  + 16 * Gg;         // 256
  float* sQs  = sBst + Ss;             // 256
  float* sBz  = sQs + Ss;              // 128
  float* sBr  = sBz + Gg;              // 128
  float* sBh  = sBr + Gg;              // 128
  float* sRe  = sBh + Gg;              // 8 (scalar + pad, keeps bf16 16B-aligned)
  __bf16* sY  = (__bf16*)(sRe + 8);    // [16,512] innovation (bf16)
  __bf16* sHx = sY + 16 * Dd;          // [16,384] concat(h|r*h, x_post) (bf16)
  char* sStage = (char*)(sHx + 16 * 384); // 16 waves * 2 * 1KB async B staging

  const int tid  = threadIdx.x;
  const int lane = tid & 31;
  const int wave = tid >> 5;           // 0..15
  const int col  = lane & 15;
  const int rb   = (lane >> 4) * 8;
  char* sb0 = sStage + wave * 2048;
  char* sb1 = sb0 + 1024;

  for (int i = tid; i < 16 * Ss; i += SCAN_THREADS) { sXe[i] = 0.0f; sP[i] = 1.0f; }
  for (int i = tid; i < 16 * Gg; i += SCAN_THREADS) sH[i] = 0.0f;
  if (tid < Ss) { sBst[tid] = b_state[tid]; sQs[tid] = qsp[tid]; }
  if (tid < Gg) { sBz[tid] = b_z[tid]; sBr[tid] = b_r[tid]; sBh[tid] = b_h[tid]; }
  if (tid == 0) sRe[0] = reff[0];
  __syncthreads();

  for (int t = 0; t < Tt; ++t) {
    const float* uT = u + (size_t)t * Dd;  // row b lives at uT + b*(Tt*Dd)

    // -- Phase 1: x_pred = x_est@A^T + u_t@W_state + b_state (16 N-tiles)
    {
      const int n0 = wave << 4;
      v8f acc = wmma_kloop(
          [&](int kb) { return frag_a_f32(sXe, (size_t)Ss, kb, lane); },
          Ab + (size_t)n0 * Ss, Ss, sb0, sb1, lane, zero8());
      acc = wmma_kloop(
          [&](int kb) { return frag_a_f32(uT, (size_t)Tt * Dd, kb, lane); },
          WstT + (size_t)n0 * Dd, Dd, sb0, sb1, lane, acc);
#pragma unroll
      for (int i = 0; i < 8; ++i)
        sXp[(rb + i) * Ss + n0 + col] = acc[i] + sBst[n0 + col];
    }
    // P_pred = clip(P + softplus(Q))
    for (int i = tid; i < 16 * Ss; i += SCAN_THREADS)
      sP[i] = clampf(sP[i] + sQs[i & (Ss - 1)], 1e-6f, 10.0f);
    __syncthreads();

    // -- Phase 2: y = clip(u_t - x_pred @ H^T)  (32 N-tiles)
    for (int tt = wave; tt < 32; tt += 16) {
      const int n0 = tt << 4;
      v8f acc = wmma_kloop(
          [&](int kb) { return frag_a_f32(sXp, (size_t)Ss, kb, lane); },
          Hb + (size_t)n0 * Ss, Ss, sb0, sb1, lane, zero8());
#pragma unroll
      for (int i = 0; i < 8; ++i) {
        int b = rb + i, d = n0 + col;
        float yv = u[((size_t)b * Tt + t) * Dd + d] - acc[i];
        sY[b * Dd + d] = (__bf16)clampf(yv, -10.0f, 10.0f);
      }
    }
    __syncthreads();

    // -- Phase 3: gain, x_post = x_pred + K*(y@H), P_post, fill hx (16 tiles)
    {
      const int n0 = wave << 4;
      v8f acc = wmma_kloop(
          [&](int kb) { return frag_a_bf16(sY, Dd, kb, lane); },
          Ht + (size_t)n0 * Dd, Dd, sb0, sb1, lane, zero8());
      const float re = sRe[0];
#pragma unroll
      for (int i = 0; i < 8; ++i) {
        int b = rb + i, s = n0 + col, idx = b * Ss + s;
        float pp = sP[idx];
        float kg = clampf(pp / (pp + re + 1e-6f), 0.0f, 1.0f);
        float xp = sXp[idx] + kg * acc[i];
        sXe[idx] = xp;                           // x_post
        sHx[b * 384 + Gg + s] = (__bf16)xp;
        sP[idx] = clampf(pp * (1.0f - kg), 1e-6f, 10.0f);
      }
    }
    for (int i = tid; i < 16 * Gg; i += SCAN_THREADS)
      sHx[(i >> 7) * 384 + (i & (Gg - 1))] = (__bf16)sH[i];
    __syncthreads();

    // -- Phase 4: z / r gates (8 tiles each; waves 0-7 -> z, 8-15 -> r)
    {
      const __bf16* W = (wave < 8) ? Wz : Wr;
      float* outb     = (wave < 8) ? sZ : sR;
      const float* bb = (wave < 8) ? sBz : sBr;
      const int n0 = (wave & 7) << 4;
      v8f acc = wmma_kloop(
          [&](int kb) { return frag_a_bf16(sHx, 384, kb, lane); },
          W + (size_t)n0 * 384, 384, sb0, sb1, lane, zero8());
#pragma unroll
      for (int i = 0; i < 8; ++i)
        outb[(rb + i) * Gg + n0 + col] = sigmoidf_(acc[i] + bb[n0 + col]);
    }
    __syncthreads();

    // -- Phase 5a: hx[:, :G] = r*h
    for (int i = tid; i < 16 * Gg; i += SCAN_THREADS)
      sHx[(i >> 7) * 384 + (i & (Gg - 1))] = (__bf16)(sR[i] * sH[i]);
    __syncthreads();

    // -- Phase 5b: hc = tanh(hx@Wh^T + b_h); h = (1-z)h + z*hc  (waves 0-7)
    if (wave < 8) {
      const int n0 = wave << 4;
      v8f acc = wmma_kloop(
          [&](int kb) { return frag_a_bf16(sHx, 384, kb, lane); },
          Wh + (size_t)n0 * 384, 384, sb0, sb1, lane, zero8());
#pragma unroll
      for (int i = 0; i < 8; ++i) {
        int b = rb + i, g = n0 + col, idx = b * Gg + g;
        float hc = tanhf(acc[i] + sBh[g]);
        float z  = sZ[idx];
        sH[idx]  = (1.0f - z) * sH[idx] + z * hc;
      }
    }
    __syncthreads();

    // -- Phase 6: x_final = x_post + h@W_out; emit xs[:,t,:]; carry (16 tiles)
    {
      const int n0 = wave << 4;
      v8f acc = wmma_kloop(
          [&](int kb) { return frag_a_f32(sH, (size_t)Gg, kb, lane); },
          WoutT + (size_t)n0 * Gg, Gg, sb0, sb1, lane, zero8());
#pragma unroll
      for (int i = 0; i < 8; ++i) {
        int b = rb + i, s = n0 + col, idx = b * Ss + s;
        float xf = sXe[idx] + acc[i];
        sXe[idx] = xf;
        xs[((size_t)b * Tt + t) * Ss + s] = xf;
      }
    }
    __syncthreads();
  }
}

// ---------------------------------------------------------------------------
// Host launcher
// ---------------------------------------------------------------------------
static inline size_t al256(size_t x) { return (x + 255) & ~(size_t)255; }

extern "C" void kernel_launch(void* const* d_in, const int* in_sizes, int n_in,
                              void* d_out, int out_size, void* d_ws, size_t ws_size,
                              hipStream_t stream) {
  (void)in_sizes; (void)n_in; (void)out_size; (void)ws_size;
  const float* x      = (const float*)d_in[0];
  const float* W_in   = (const float*)d_in[1];
  const float* b_in   = (const float*)d_in[2];
  const float* W_st   = (const float*)d_in[3];
  const float* b_st   = (const float*)d_in[4];
  const float* A      = (const float*)d_in[5];
  const float* H      = (const float*)d_in[6];
  const float* Q      = (const float*)d_in[7];
  const float* R      = (const float*)d_in[8];
  const float* W_z    = (const float*)d_in[9];
  const float* W_r    = (const float*)d_in[10];
  const float* W_h    = (const float*)d_in[11];
  const float* b_z    = (const float*)d_in[12];
  const float* b_r    = (const float*)d_in[13];
  const float* b_h    = (const float*)d_in[14];
  const float* W_out  = (const float*)d_in[15];
  const float* W_outp = (const float*)d_in[16];
  const float* b_outp = (const float*)d_in[17];
  float* out = (float*)d_out;

  const size_t BT = (size_t)Bb * Tt;
  char* ws = (char*)d_ws;
  size_t off = 0;
  float* u    = (float*)(ws + off); off = al256(off + BT * Dd * 4);
  float* xs   = (float*)(ws + off); off = al256(off + BT * Ss * 4);
  float* obs  = (float*)(ws + off); off = al256(off + BT * Dd * 4);
  float* qsp  = (float*)(ws + off); off = al256(off + Ss * 4);
  float* reff = (float*)(ws + off); off = al256(off + 256);
  __bf16* WinT   = (__bf16*)(ws + off); off = al256(off + (size_t)Dd * Ee * 2);
  __bf16* Ab     = (__bf16*)(ws + off); off = al256(off + (size_t)Ss * Ss * 2);
  __bf16* WstT   = (__bf16*)(ws + off); off = al256(off + (size_t)Ss * Dd * 2);
  __bf16* Hb     = (__bf16*)(ws + off); off = al256(off + (size_t)Dd * Ss * 2);
  __bf16* Ht     = (__bf16*)(ws + off); off = al256(off + (size_t)Ss * Dd * 2);
  __bf16* Wzb    = (__bf16*)(ws + off); off = al256(off + (size_t)Gg * 384 * 2);
  __bf16* Wrb    = (__bf16*)(ws + off); off = al256(off + (size_t)Gg * 384 * 2);
  __bf16* Whb    = (__bf16*)(ws + off); off = al256(off + (size_t)Gg * 384 * 2);
  __bf16* WoutT  = (__bf16*)(ws + off); off = al256(off + (size_t)Ss * Gg * 2);
  __bf16* WoutpT = (__bf16*)(ws + off); off = al256(off + (size_t)Ee * Dd * 2);

  // softplus(Q), mean(softplus(R))
  k_prep<<<1, 512, 0, stream>>>(Q, R, qsp, reff);

  // weight casts / transposes (stored [N][K] bf16 for the B-fragment layout)
  auto grid1 = [](int n) { return dim3((n + 255) / 256); };
  k_cast_t<<<grid1(Ee * Dd), 256, 0, stream>>>(W_in, WinT, Ee, Dd);     // [D][E]
  k_cast  <<<grid1(Ss * Ss), 256, 0, stream>>>(A, Ab, Ss * Ss);         // A row-major == [N=S][K=S] of A^T
  k_cast_t<<<grid1(Dd * Ss), 256, 0, stream>>>(W_st, WstT, Dd, Ss);     // [S][D]
  k_cast  <<<grid1(Dd * Ss), 256, 0, stream>>>(H, Hb, Dd * Ss);         // H row-major == [N=D][K=S] of H^T
  k_cast_t<<<grid1(Dd * Ss), 256, 0, stream>>>(H, Ht, Dd, Ss);          // [S][D] for y@H
  k_cast  <<<grid1(Gg * 384), 256, 0, stream>>>(W_z, Wzb, Gg * 384);    // row-major == [N=G][K=384] of W^T
  k_cast  <<<grid1(Gg * 384), 256, 0, stream>>>(W_r, Wrb, Gg * 384);
  k_cast  <<<grid1(Gg * 384), 256, 0, stream>>>(W_h, Whb, Gg * 384);
  k_cast_t<<<grid1(Gg * Ss), 256, 0, stream>>>(W_out, WoutT, Gg, Ss);   // [S][G]
  k_cast_t<<<grid1(Dd * Ee), 256, 0, stream>>>(W_outp, WoutpT, Dd, Ee); // [E][D]

  // GEMM1: u = gelu(x @ W_in + b_in)   [16384,1024]x[1024,512]
  k_gemm_bf16<<<dim3(Dd / 128, (int)(BT / 16)), 256, 0, stream>>>(
      x, WinT, b_in, nullptr, u, (int)BT, Dd, Ee, 1);

  // Recurrent Kalman+GRU scan: one WGP-resident workgroup, 16 waves
  k_scan<<<1, SCAN_THREADS, SCAN_SMEM, stream>>>(
      u, xs, Ab, WstT, Hb, Ht, Wzb, Wrb, Whb, WoutT,
      b_st, b_z, b_r, b_h, qsp, reff);

  // GEMM3: obs = xs @ H^T   [16384,256]x[256,512]
  k_gemm_bf16<<<dim3(Dd / 128, (int)(BT / 16)), 256, 0, stream>>>(
      xs, Hb, nullptr, nullptr, obs, (int)BT, Dd, Ss, 0);

  // GEMM4: out = obs @ W_outp + b_outp + x   [16384,512]x[512,1024]
  k_gemm_bf16<<<dim3(Ee / 128, (int)(BT / 16)), 256, 0, stream>>>(
      obs, WoutpT, b_outp, x, out, (int)BT, Ee, Dd, 0);
}
